// MoEBlock_87119116632099
// MI455X (gfx1250) — compile-verified, matrix operand order checked by reading, not run on hardware
//
#include <hip/hip_runtime.h>
#include <hip/hip_bf16.h>
#include <math.h>

// ---------------------------------------------------------------------------
// MoE block for MI455X (gfx1250, wave32, WMMA).
//   B=8 T=4096 D=512 H=2048 E=8 TOPK=2, capacity = Bt = 32768.
//   out[t] = wy[t] * expert_ffn_{imax[t]}(x[t]) + wx[t] * x[t]
// Weights are pre-converted to bf16 and pre-transposed (B-operand major) once
// per launch; the FFN hot loop is then pure b128 loads + WMMA.
// ---------------------------------------------------------------------------

#define BT     32768
#define DD     512
#define HH     2048
#define EE     8
#define SX     528   // lx row stride (bf16 elems): 1056B, 16B-aligned rows
#define SHH    72    // lh row stride (bf16 elems): 144B, 16B-aligned rows
#define NTILE  32    // tokens per ffn block
#define MAXTILES 1032

typedef __attribute__((ext_vector_type(16))) __bf16 v16bf;
typedef __attribute__((ext_vector_type(8)))  float  v8f;

union Frag {
    v16bf v;
    unsigned int u[8];
    uint4 q[2];
};

__device__ __forceinline__ unsigned short f2bf(float f) {
    unsigned int u = __float_as_uint(f);
    u += 0x7fffu + ((u >> 16) & 1u);        // round-to-nearest-even
    return (unsigned short)(u >> 16);
}

// A fragment (16 x K=32, bf16) from LDS, row-major, stride in elems (16B-mult).
// Lane l<16: row m0+l, K {k0..k0+7, k0+16..k0+23}; lane>=16: +8 K shift.
__device__ __forceinline__ void load_a_lds(const unsigned short* src, int stride,
                                           int m0, int k0, Frag& f) {
    int lane = threadIdx.x & 31;
    int row  = m0 + (lane & 15);
    int kb   = k0 + ((lane >> 4) << 3);
    const unsigned short* p = src + row * stride + kb;
    f.q[0] = *(const uint4*)(p);
    f.q[1] = *(const uint4*)(p + 16);
}

// B fragment (K=32 x 16, bf16) from a pre-transposed bf16 weight in global:
// WT[e][n][k], leading dim ldk. Lane l<16: col n0+l, K {k0..k0+7,k0+16..k0+23};
// lane>=16: +8 K shift. Two global_load_b128 per lane, no conversion.
__device__ __forceinline__ void load_b_gbf(const unsigned short* WT, int ldk,
                                           int k0, int n0, Frag& f) {
    int lane = threadIdx.x & 31;
    int col  = n0 + (lane & 15);
    int kb   = k0 + ((lane >> 4) << 3);
    const unsigned short* p = WT + (size_t)col * ldk + kb;
    f.q[0] = *(const uint4*)(p);
    f.q[1] = *(const uint4*)(p + 16);
}

__device__ __forceinline__ v8f bwmma(const Frag& a, const Frag& b, v8f c) {
    return __builtin_amdgcn_wmma_f32_16x16x32_bf16(
        false, a.v, false, b.v, (short)0, c, false, false);
}

// ---------------------------------------------------------------------------
// Transpose + f32->bf16 convert: src[e][k][n] (R x C) -> dst[e][n][k] (C x R).
// 32x32 tiles through LDS; packed bf16x2 stores.
// ---------------------------------------------------------------------------
__global__ __launch_bounds__(256) void cvt_t_kernel(
    const float* __restrict__ src, unsigned short* __restrict__ dst,
    int R, int C)
{
    __shared__ float tile[32][33];
    int tilesR = R >> 5, tilesC = C >> 5;
    int b  = blockIdx.x;
    int tk = b % tilesR; b /= tilesR;
    int tn = b % tilesC; b /= tilesC;
    int e  = b;

    const float* s = src + ((size_t)e * R + tk * 32) * C + tn * 32;
    int lr = threadIdx.x >> 5, lc = threadIdx.x & 31;
#pragma unroll
    for (int i = 0; i < 4; ++i) {
        int rr = i * 8 + lr;
        tile[rr][lc] = s[(size_t)rr * C + lc];
    }
    __syncthreads();

    unsigned short* d = dst + ((size_t)e * C + tn * 32) * R + tk * 32;
#pragma unroll
    for (int i = 0; i < 2; ++i) {
        int idx = threadIdx.x + i * 256;    // 0..511
        int n = idx >> 4, kp = idx & 15;    // bf16-pair index along k
        unsigned int lo = f2bf(tile[2 * kp][n]);
        unsigned int hi = f2bf(tile[2 * kp + 1][n]);
        *(unsigned int*)(d + (size_t)n * R + 2 * kp) = lo | (hi << 16);
    }
}

// ---------------------------------------------------------------------------
// Router: one wave per token. logits = x[t] @ Wr + br ; top-2 ; softmax.
// ---------------------------------------------------------------------------
__global__ __launch_bounds__(256) void router_kernel(
    const float* __restrict__ x, const float* __restrict__ Wr,
    const float* __restrict__ br,
    int* __restrict__ idx0, int* __restrict__ idx1,
    float* __restrict__ s0, float* __restrict__ s1)
{
    int wid  = threadIdx.x >> 5;
    int lane = threadIdx.x & 31;
    int t    = blockIdx.x * 8 + wid;
    const float* xr = x + (size_t)t * DD;

    float acc[EE];
#pragma unroll
    for (int e = 0; e < EE; ++e) acc[e] = 0.f;

    for (int j = 0; j < DD / 32; ++j) {
        int d = j * 32 + lane;                 // coalesced
        float xv = xr[d];
        const float* wr = Wr + d * EE;
#pragma unroll
        for (int e = 0; e < EE; ++e) acc[e] += xv * wr[e];
    }
#pragma unroll
    for (int s = 16; s >= 1; s >>= 1) {
#pragma unroll
        for (int e = 0; e < EE; ++e)
            acc[e] += __shfl_xor(acc[e], s, 32);
    }
    if (lane == 0) {
        float v0 = -1e30f, v1 = -1e30f;
        int   i0 = 0, i1 = 0;
#pragma unroll
        for (int e = 0; e < EE; ++e) {
            float v = acc[e] + br[e];
            if (v > v0)      { v1 = v0; i1 = i0; v0 = v; i0 = e; }
            else if (v > v1) { v1 = v;  i1 = e; }
        }
        float ex = expf(v1 - v0);
        float dn = 1.f + ex;
        idx0[t] = i0; idx1[t] = i1;
        s0[t] = 1.f / dn; s1[t] = ex / dn;
    }
}

// ---------------------------------------------------------------------------
// Capacity scan: exact double-cumsum (inclusive) of one-hot assignments.
//   pos[t,0,e] = #{t'<=t : idx0[t']==e}                     (slot-0 only)
//   pos[t,1,e] = #{t'<=t : idx0[t']==e} + #{t'<=t : idx1[t']==e}
// kept iff pos < capacity (= BT). Emits wy/wx and per-expert token buckets.
// ---------------------------------------------------------------------------
__global__ __launch_bounds__(256) void scan_kernel(
    const int* __restrict__ idx0, const int* __restrict__ idx1,
    const float* __restrict__ s0, const float* __restrict__ s1,
    float* __restrict__ wy, float* __restrict__ wx,
    int* __restrict__ cnt, int* __restrict__ bucket)
{
    __shared__ int l0[256][EE];
    __shared__ int l01[256][EE];
    int tid  = threadIdx.x;
    int base = tid * (BT / 256);

    int c0[EE], c01[EE];
#pragma unroll
    for (int e = 0; e < EE; ++e) { c0[e] = 0; c01[e] = 0; }
    for (int i = 0; i < BT / 256; ++i) {
        int t = base + i;
        int e0 = idx0[t], e1 = idx1[t];
        c0[e0]++; c01[e0]++; c01[e1]++;
    }
#pragma unroll
    for (int e = 0; e < EE; ++e) { l0[tid][e] = c0[e]; l01[tid][e] = c01[e]; }
    __syncthreads();

    int p0[EE], p01[EE];
#pragma unroll
    for (int e = 0; e < EE; ++e) { p0[e] = 0; p01[e] = 0; }
    for (int t = 0; t < tid; ++t) {
#pragma unroll
        for (int e = 0; e < EE; ++e) { p0[e] += l0[t][e]; p01[e] += l01[t][e]; }
    }

    for (int i = 0; i < BT / 256; ++i) {
        int t = base + i;
        int e0 = idx0[t], e1 = idx1[t];
        p0[e0]++;  int pos0 = p0[e0];
        p01[e0]++; p01[e1]++; int pos1 = p01[e1];
        bool k0 = pos0 < BT;
        bool k1 = pos1 < BT;
        int a0 = k0 ? e0 : 0;
        int a1 = k1 ? e1 : 0;
        int im = a0 > a1 ? a0 : a1;
        float sv0 = s0[t], sv1 = s1[t];
        wy[t] = (k0 ? sv0 : 0.f) + (k1 ? sv1 : 0.f);
        wx[t] = (k0 ? 0.f : sv0) + (k1 ? 0.f : sv1);
        int slot = atomicAdd(&cnt[im], 1);
        bucket[im * BT + slot] = t;
    }
}

// ---------------------------------------------------------------------------
// Expert FFN on 32-token tiles with bf16 WMMA.
//   h = gelu((x@W2) * (x@W1));  y = h @ W3;  out = wy*y + wx*x
// Per round (64 H-cols): phase1 builds h tile in LDS (8 waves: mi x nj),
// phase2 accumulates y (each wave owns 64 of 512 output cols).
// ---------------------------------------------------------------------------
__global__ __launch_bounds__(256) void ffn_kernel(
    const float* __restrict__ x,
    const unsigned short* __restrict__ W1b,   // [E][H][D] bf16
    const unsigned short* __restrict__ W2b,   // [E][H][D] bf16
    const unsigned short* __restrict__ W3b,   // [E][D][H] bf16
    const int* __restrict__ cnt, const int* __restrict__ bucket,
    const float* __restrict__ wy, const float* __restrict__ wx,
    float* __restrict__ out)
{
    __shared__ __align__(16) unsigned short lx[NTILE * SX];   // x tile, bf16
    __shared__ __align__(16) unsigned short lh[NTILE * SHH];  // h round tile
    __shared__ int stok[NTILE];

    // Map block -> (expert, tile) from the bucket counts.
    int e = -1, tile = 0, acc_t = 0;
#pragma unroll
    for (int i = 0; i < EE; ++i) {
        int te = (cnt[i] + NTILE - 1) >> 5;
        if (e < 0 && (int)blockIdx.x < acc_t + te) { e = i; tile = blockIdx.x - acc_t; }
        acc_t += te;
    }
    if (e < 0) return;
    int nt = cnt[e] - tile * NTILE;
    if (nt > NTILE) nt = NTILE;

    const int* bk = bucket + e * BT + tile * NTILE;
    if (threadIdx.x < NTILE)
        stok[threadIdx.x] = (threadIdx.x < nt) ? bk[threadIdx.x] : 0;
    __syncthreads();

    // Stage x tile (f32 -> bf16 pairs) into LDS; zero-pad invalid rows.
    for (int it = 0; it < (NTILE * DD / 2) / 256; ++it) {
        int idx = threadIdx.x + it * 256;    // bf16-pair index
        int row = idx >> 8, cp = idx & 255;
        float v0 = 0.f, v1 = 0.f;
        if (row < nt) {
            const float* xp = x + (size_t)stok[row] * DD + 2 * cp;
            v0 = xp[0]; v1 = xp[1];
        }
        *(unsigned int*)&lx[row * SX + 2 * cp] =
            (unsigned int)f2bf(v0) | ((unsigned int)f2bf(v1) << 16);
    }
    __syncthreads();

    const unsigned short* W1e = W1b + (size_t)e * DD * HH;
    const unsigned short* W2e = W2b + (size_t)e * DD * HH;
    const unsigned short* W3e = W3b + (size_t)e * HH * DD;

    int w    = threadIdx.x >> 5;
    int lane = threadIdx.x & 31;
    int mi   = w & 1;       // phase-1 M sub-tile
    int nj   = w >> 1;      // phase-1 N sub-tile within 64-col round

    v8f accy[2][4];
#pragma unroll
    for (int a = 0; a < 2; ++a)
#pragma unroll
        for (int b = 0; b < 4; ++b) accy[a][b] = (v8f){};

    for (int r = 0; r < HH / 64; ++r) {
        // ---- phase 1: h[32 x 64] for this round ----
        int n0 = r * 64 + nj * 16;
        // prefetch next round's weight rows (global_prefetch_b8)
        if (r + 1 < HH / 64) {
            __builtin_prefetch(W1e + (size_t)(n0 + 64 + (lane & 15)) * DD, 0, 1);
            __builtin_prefetch(W2e + (size_t)(n0 + 64 + (lane & 15)) * DD, 0, 1);
        }
        v8f a1 = (v8f){}, a2 = (v8f){};
        for (int ks = 0; ks < DD / 32; ++ks) {
            Frag fa, fb1, fb2;
            load_a_lds(lx, SX, mi * 16, ks * 32, fa);
            load_b_gbf(W1e, DD, ks * 32, n0, fb1);
            load_b_gbf(W2e, DD, ks * 32, n0, fb2);
            a1 = bwmma(fa, fb1, a1);
            a2 = bwmma(fa, fb2, a2);
        }
        int nl = nj * 16 + (lane & 15);
        int mb = mi * 16 + ((lane >> 4) << 3);
#pragma unroll
        for (int rr = 0; rr < 8; ++rr) {
            float v = a1[rr] * a2[rr];
            float g = 0.5f * v * (1.f + erff(v * 0.70710678118654752f));
            lh[(mb + rr) * SHH + nl] = f2bf(g);
        }
        __syncthreads();

        // ---- phase 2: y[:, w*64 .. w*64+64) += h @ W3[round rows] ----
        for (int ks = 0; ks < 2; ++ks) {
            Frag fa0, fa1f, fb;
            load_a_lds(lh, SHH, 0,  ks * 32, fa0);
            load_a_lds(lh, SHH, 16, ks * 32, fa1f);
#pragma unroll
            for (int nj2 = 0; nj2 < 4; ++nj2) {
                load_b_gbf(W3e, HH, r * 64 + ks * 32, w * 64 + nj2 * 16, fb);
                accy[0][nj2] = bwmma(fa0,  fb, accy[0][nj2]);
                accy[1][nj2] = bwmma(fa1f, fb, accy[1][nj2]);
            }
        }
        __syncthreads();
    }

    // ---- store: out = wy*y + wx*x ----
#pragma unroll
    for (int mi2 = 0; mi2 < 2; ++mi2) {
#pragma unroll
        for (int nj2 = 0; nj2 < 4; ++nj2) {
            int n     = w * 64 + nj2 * 16 + (lane & 15);
            int mbase = mi2 * 16 + ((lane >> 4) << 3);
#pragma unroll
            for (int rr = 0; rr < 8; ++rr) {
                int m = mbase + rr;
                if (m < nt) {
                    int tok = stok[m];
                    float val = wy[tok] * accy[mi2][nj2][rr]
                              + wx[tok] * x[(size_t)tok * DD + n];
                    out[(size_t)tok * DD + n] = val;
                }
            }
        }
    }
}

// ---------------------------------------------------------------------------
// Launch
// ---------------------------------------------------------------------------
extern "C" void kernel_launch(void* const* d_in, const int* in_sizes, int n_in,
                              void* d_out, int out_size, void* d_ws, size_t ws_size,
                              hipStream_t stream) {
    const float* x  = (const float*)d_in[0];
    const float* Wr = (const float*)d_in[1];
    const float* br = (const float*)d_in[2];
    const float* W1 = (const float*)d_in[3];
    const float* W2 = (const float*)d_in[4];
    const float* W3 = (const float*)d_in[5];
    float* out = (float*)d_out;

    char* ws = (char*)d_ws;
    // workspace layout (bytes)
    int*   cnt    = (int*)  (ws);                       // 8 ints (+pad to 256)
    float* wy     = (float*)(ws + 256);                 // BT
    float* wx     = (float*)(ws + 256 + 131072);        // BT
    int*   idx0   = (int*)  (ws + 256 + 2 * 131072);    // BT
    int*   idx1   = (int*)  (ws + 256 + 3 * 131072);    // BT
    float* s0     = (float*)(ws + 256 + 4 * 131072);    // BT
    float* s1     = (float*)(ws + 256 + 5 * 131072);    // BT
    int*   bucket = (int*)  (ws + 256 + 6 * 131072);    // EE*BT ints
    const size_t WB = (size_t)EE * DD * HH * 2;         // 16 MB per weight
    unsigned short* W1b = (unsigned short*)(ws + (2u << 20));
    unsigned short* W2b = (unsigned short*)(ws + (2u << 20) + WB);
    unsigned short* W3b = (unsigned short*)(ws + (2u << 20) + 2 * WB);

    hipMemsetAsync(cnt, 0, EE * sizeof(int), stream);

    // Pre-convert weights: bf16, B-operand major.
    cvt_t_kernel<<<(DD / 32) * (HH / 32) * EE, 256, 0, stream>>>(W1, W1b, DD, HH);
    cvt_t_kernel<<<(DD / 32) * (HH / 32) * EE, 256, 0, stream>>>(W2, W2b, DD, HH);
    cvt_t_kernel<<<(HH / 32) * (DD / 32) * EE, 256, 0, stream>>>(W3, W3b, HH, DD);

    router_kernel<<<BT / 8, 256, 0, stream>>>(x, Wr, br, idx0, idx1, s0, s1);
    scan_kernel<<<1, 256, 0, stream>>>(idx0, idx1, s0, s1, wy, wx, cnt, bucket);
    ffn_kernel<<<MAXTILES, 256, 0, stream>>>(x, W1b, W2b, W3b, cnt, bucket, wy, wx, out);
}
